// NeuronMiniMaxM2DecoderLayer_79989470921200
// MI455X (gfx1250) — matrix operand order repaired and checked by reading, not compiled, not run on hardware
//
#include <hip/hip_runtime.h>

// ---------------------------------------------------------------------------
// MiniMax-M2 decoder layer for MI455X (gfx1250, wave32, WMMA).
// bf16 inputs + f32 WMMA accumulation; weights pre-converted to bf16 [N,K]
// so all WMMA fragments are contiguous vector loads; 4 M-tiles per wave to
// amortize weight bandwidth; top-2-only MoE via expert gather.
// ---------------------------------------------------------------------------

typedef __attribute__((ext_vector_type(8)))  __bf16 v8bf;
typedef __attribute__((ext_vector_type(16))) __bf16 v16bf;
typedef __attribute__((ext_vector_type(8)))  float  v8f;

constexpr int kS   = 1024;
constexpr int kH   = 2048;
constexpr int kNH  = 16;
constexpr int kNKV = 4;
constexpr int kHD  = 128;
constexpr int kE   = 8;
constexpr int kI   = 1024;
constexpr int kTOPK = 2;
constexpr float kEPS = 1e-6f;
constexpr float kSCALE = 0.08838834764831843f; // 1/sqrt(128)

__device__ inline v8f vzero8() {
  v8f z = {0.f, 0.f, 0.f, 0.f, 0.f, 0.f, 0.f, 0.f};
  return z;
}

__device__ inline v8f wmma_bf16(v16bf a, v16bf b, v8f c) {
  return __builtin_amdgcn_wmma_f32_16x16x32_bf16(
      /*neg_a=*/false, a, /*neg_b=*/false, b,
      /*c_mod=*/(short)0, c, /*reuse_a=*/false, /*reuse_b=*/false);
}

// A-fragment (16x32 bf16, row-major source): lane m = lane&15, hh = lane>>4.
// element e -> k = 16*hh+e. Two contiguous 16B runs.
__device__ inline v16bf load_a_bf(const __bf16* Arow, int k0, int hh) {
  v8bf lo = *(const v8bf*)(Arow + k0 + 16 * hh);
  v8bf hi = *(const v8bf*)(Arow + k0 + 16 * hh + 8);
  v16bf a;
#pragma unroll
  for (int i = 0; i < 8; ++i) { a[i] = lo[i]; a[i + 8] = hi[i]; }
  return a;
}

// B-fragment where the 16 K-elements per lane are contiguous bf16 in memory
// (element e -> k = 16*hh + e). Used with Bt stored as [N, K].
__device__ inline v16bf load_b_contig(const __bf16* p) {
  v8bf lo = *(const v8bf*)p;
  v8bf hi = *(const v8bf*)(p + 8);
  v16bf b;
#pragma unroll
  for (int i = 0; i < 8; ++i) { b[i] = lo[i]; b[i + 8] = hi[i]; }
  return b;
}

// ----------------- weight transpose + f32 -> bf16 conversion ---------------
// src: f32 [batch, K, N] row-major; dst: bf16 [batch, N, K].
__global__ void k_wt(const float* __restrict__ B, __bf16* __restrict__ Bt, int K, int N) {
  __shared__ float tile[32][33];
  const float* Bb = B + (size_t)blockIdx.z * K * N;
  __bf16* Btb = Bt + (size_t)blockIdx.z * K * N;
  int kb = blockIdx.y * 32, nb = blockIdx.x * 32;
  int tx = threadIdx.x, ty = threadIdx.y; // 32 x 8
  for (int i = ty; i < 32; i += 8)
    tile[i][tx] = Bb[(size_t)(kb + i) * N + nb + tx];
  __syncthreads();
  for (int i = ty; i < 32; i += 8)
    Btb[(size_t)(nb + i) * K + kb + tx] = (__bf16)tile[tx][i];
}

// ------------------------------- RMSNorm -----------------------------------
__global__ void k_rmsnorm(const float* __restrict__ x, const float* __restrict__ w,
                          float* __restrict__ outf, __bf16* __restrict__ outb, int n) {
  int row = blockIdx.x;
  int tix = threadIdx.x;
  const float* xr = x + (size_t)row * n;
  float ss = 0.f;
  for (int i = tix; i < n; i += blockDim.x) { float v = xr[i]; ss += v * v; }
  __shared__ float red[256];
  red[tix] = ss;
  __syncthreads();
  for (int s = 128; s > 0; s >>= 1) {
    if (tix < s) red[tix] += red[tix + s];
    __syncthreads();
  }
  float scale = rsqrtf(red[0] / (float)n + kEPS);
  for (int i = tix; i < n; i += blockDim.x) {
    float v = xr[i] * scale * w[i];
    if (outf) outf[(size_t)row * n + i] = v;
    if (outb) outb[(size_t)row * n + i] = (__bf16)v;
  }
}

// ------------------------------ Generic GEMM -------------------------------
// C[M,N] = A_bf16[M,K] * Bt_bf16[N,K]^T (+resid), f32 out.
// One wave computes MT vertical 16x16 tiles sharing each B fragment.
template <int MT, bool RES>
__global__ __launch_bounds__(32)
void k_gemm_t(const __bf16* __restrict__ A, const __bf16* __restrict__ Bt,
              const float* __restrict__ resid, float* __restrict__ Cf, int N, int K) {
  int n0 = blockIdx.x * 16, m0 = blockIdx.y * 16 * MT;
  int lane = threadIdx.x, mrow = lane & 15, hh = lane >> 4;
  const __bf16* Arow[MT];
#pragma unroll
  for (int t = 0; t < MT; ++t) Arow[t] = A + (size_t)(m0 + t * 16 + mrow) * K;
  const __bf16* Bp = Bt + (size_t)(n0 + mrow) * K;
  v8f acc[MT];
#pragma unroll
  for (int t = 0; t < MT; ++t) acc[t] = vzero8();
  for (int k0 = 0; k0 < K; k0 += 32) {
    __builtin_prefetch((const void*)(Bp + k0 + 512), 0, 1);
    v16bf b = load_b_contig(Bp + k0 + 16 * hh);
#pragma unroll
    for (int t = 0; t < MT; ++t) {
      v16bf a = load_a_bf(Arow[t], k0, hh);
      acc[t] = wmma_bf16(a, b, acc[t]);
    }
  }
#pragma unroll
  for (int t = 0; t < MT; ++t)
#pragma unroll
    for (int v = 0; v < 8; ++v) {
      int m = m0 + t * 16 + v + 8 * hh;
      int n = n0 + mrow;
      float val = acc[t][v];
      if (RES) val += resid[(size_t)m * N + n];
      Cf[(size_t)m * N + n] = val;
    }
}

// ----------------------- RoPE -> head-major bf16 ---------------------------
// in: f32 [S, nheads*HD]; out: bf16 [nheads, S, HD].
__global__ void k_rope(const float* __restrict__ in, const int* __restrict__ pos,
                       __bf16* __restrict__ out, int nheads) {
  int s = blockIdx.x, hIdx = blockIdx.y, d = threadIdx.x;
  size_t base = (size_t)s * nheads * kHD + (size_t)hIdx * kHD;
  float v = in[base + d];
  if (d < 64) {
    int p = pos[s];
    int j = (d < 32) ? d : d - 32;
    float inv = expf(-((2.0f * j) / 64.0f) * 13.815510557964274f); // ln(1e6)
    float ang = (float)p * inv;
    float c = cosf(ang), sn = sinf(ang);
    float other = in[base + ((d < 32) ? d + 32 : d - 32)];
    v = (d < 32) ? (v * c - other * sn) : (v * c + other * sn);
  }
  out[((size_t)hIdx * kS + s) * kHD + d] = (__bf16)v;
}

// V: f32 [S, NKV*HD] -> bf16 transposed [NKV, HD, S] (PV B-fragments contiguous).
__global__ void k_vt(const float* __restrict__ in, __bf16* __restrict__ out) {
  int s = blockIdx.x, hIdx = blockIdx.y, d = threadIdx.x;
  out[((size_t)hIdx * kHD + d) * kS + s] =
      (__bf16)in[(size_t)s * kNKV * kHD + (size_t)hIdx * kHD + d];
}

// ----------------------- Flash attention (causal GQA) ----------------------
// Q: [NH,S,HD] bf16, K: [NKV,S,HD] bf16, Vt: [NKV,HD,S] bf16, O: [S,NH*HD] bf16.
__global__ __launch_bounds__(32)
void k_attn(const __bf16* __restrict__ Q, const __bf16* __restrict__ K,
            const __bf16* __restrict__ Vt, __bf16* __restrict__ O) {
  int qt = blockIdx.x, hq = blockIdx.y;
  int kvh = hq >> 2;
  int qb = qt * 16;
  int lane = threadIdx.x, mrow = lane & 15, hh = lane >> 4;
  const __bf16* Qh = Q + ((size_t)hq * kS + qb) * kHD;
  const __bf16* Kh = K + (size_t)kvh * kS * kHD;
  const __bf16* Vh = Vt + (size_t)kvh * kHD * kS;

  v16bf aq[4];
#pragma unroll
  for (int c = 0; c < 4; ++c) aq[c] = load_a_bf(Qh + (size_t)mrow * kHD, 32 * c, hh);

  v8f acc[8];
#pragma unroll
  for (int c = 0; c < 8; ++c) acc[c] = vzero8();
  float mrun[8], lrun[8];
#pragma unroll
  for (int v = 0; v < 8; ++v) { mrun[v] = -1e30f; lrun[v] = 0.f; }

  __shared__ float P[16][36];

  for (int kb = 0; kb <= qb + 15; kb += 32) {
    v8f s0 = vzero8(), s1 = vzero8();
#pragma unroll
    for (int c = 0; c < 4; ++c) {
      v16bf b0 = load_b_contig(Kh + (size_t)(kb + mrow) * kHD + 32 * c + 16 * hh);
      s0 = wmma_bf16(aq[c], b0, s0);
      v16bf b1 = load_b_contig(Kh + (size_t)(kb + 16 + mrow) * kHD + 32 * c + 16 * hh);
      s1 = wmma_bf16(aq[c], b1, s1);
    }
    float p0v[8], p1v[8], alph[8];
#pragma unroll
    for (int v = 0; v < 8; ++v) {
      int qrow = qb + v + 8 * hh;
      int k0i = kb + mrow, k1i = kb + 16 + mrow;
      float x0 = (k0i <= qrow) ? s0[v] * kSCALE : -1e30f;
      float x1 = (k1i <= qrow) ? s1[v] * kSCALE : -1e30f;
      float t = fmaxf(x0, x1);
      t = fmaxf(t, __shfl_xor(t, 1, 32));
      t = fmaxf(t, __shfl_xor(t, 2, 32));
      t = fmaxf(t, __shfl_xor(t, 4, 32));
      t = fmaxf(t, __shfl_xor(t, 8, 32));
      float mnew = fmaxf(mrun[v], t);
      alph[v] = expf(mrun[v] - mnew);
      float p0 = expf(x0 - mnew);
      float p1 = expf(x1 - mnew);
      float ls = p0 + p1;
      ls += __shfl_xor(ls, 1, 32);
      ls += __shfl_xor(ls, 2, 32);
      ls += __shfl_xor(ls, 4, 32);
      ls += __shfl_xor(ls, 8, 32);
      lrun[v] = lrun[v] * alph[v] + ls;
      mrun[v] = mnew;
      p0v[v] = p0;
      p1v[v] = p1;
    }
#pragma unroll
    for (int c = 0; c < 8; ++c)
#pragma unroll
      for (int v = 0; v < 8; ++v) acc[c][v] *= alph[v];

    // Stage P (16x32 f32) through LDS to convert C-layout -> A-layout.
#pragma unroll
    for (int v = 0; v < 8; ++v) {
      P[v + 8 * hh][mrow] = p0v[v];
      P[v + 8 * hh][16 + mrow] = p1v[v];
    }
    __syncthreads();
    v16bf ap;
#pragma unroll
    for (int e = 0; e < 8; ++e) {
      ap[e] = (__bf16)P[mrow][16 * hh + e];
      ap[8 + e] = (__bf16)P[mrow][16 * hh + 8 + e];
    }
#pragma unroll
    for (int c = 0; c < 8; ++c) {
      v16bf bv = load_b_contig(Vh + (size_t)(16 * c + mrow) * kS + kb + 16 * hh);
      acc[c] = wmma_bf16(ap, bv, acc[c]);
    }
    __syncthreads();
  }

#pragma unroll
  for (int c = 0; c < 8; ++c)
#pragma unroll
    for (int v = 0; v < 8; ++v) {
      float o = acc[c][v] / lrun[v];
      O[(size_t)(qb + v + 8 * hh) * (kNH * kHD) + (size_t)hq * kHD + 16 * c + mrow] = (__bf16)o;
    }
}

// ------------------------------- Router ------------------------------------
__global__ void k_router(const float* __restrict__ t, const float* __restrict__ rw,
                         const float* __restrict__ rb, int* __restrict__ tid_out,
                         float* __restrict__ tw_out) {
  int row = blockIdx.x, tix = threadIdx.x;
  const float* tr = t + (size_t)row * kH;
  float acc[kE];
#pragma unroll
  for (int e = 0; e < kE; ++e) acc[e] = 0.f;
  for (int h = tix; h < kH; h += 256) {
    float x = tr[h];
#pragma unroll
    for (int e = 0; e < kE; ++e) acc[e] += x * rw[h * kE + e];
  }
  __shared__ float red[256];
  __shared__ float logits[kE];
  for (int e = 0; e < kE; ++e) {
    red[tix] = acc[e];
    __syncthreads();
    for (int s = 128; s > 0; s >>= 1) {
      if (tix < s) red[tix] += red[tix + s];
      __syncthreads();
    }
    if (tix == 0) logits[e] = red[0];
    __syncthreads();
  }
  if (tix == 0) {
    float sc[kE], val[kE];
#pragma unroll
    for (int e = 0; e < kE; ++e) {
      sc[e] = 1.f / (1.f + expf(-logits[e]));
      val[e] = sc[e] + rb[e];
    }
    int i0 = 0;
    for (int e = 1; e < kE; ++e) if (val[e] > val[i0]) i0 = e;
    int i1 = -1;
    for (int e = 0; e < kE; ++e) {
      if (e == i0) continue;
      if (i1 < 0 || val[e] > val[i1]) i1 = e;
    }
    float a0 = sc[i0], a1 = sc[i1], s = a0 + a1;
    tid_out[row * 2 + 0] = i0;
    tid_out[row * 2 + 1] = i1;
    tw_out[row * 2 + 0] = a0 / s;
    tw_out[row * 2 + 1] = a1 / s;
  }
}

// ------------------------ MoE token gather machinery -----------------------
__global__ void k_zero_counts(int* counts) {
  if (threadIdx.x < kE) counts[threadIdx.x] = 0;
}
__global__ void k_assign(const int* __restrict__ topk_id, int* counts, int* posArr) {
  int i = blockIdx.x * blockDim.x + threadIdx.x;
  if (i >= kS * kTOPK) return;
  posArr[i] = atomicAdd(&counts[topk_id[i]], 1);
}
__global__ void k_offsets(const int* counts, int* off) {
  if (threadIdx.x == 0 && blockIdx.x == 0) {
    int s = 0;
    for (int e = 0; e < kE; ++e) { off[e] = s; s += counts[e]; }
    off[kE] = s;
  }
}
__global__ void k_scatter(const int* __restrict__ topk_id, const float* __restrict__ topk_w,
                          const int* __restrict__ posArr, const int* __restrict__ off,
                          int* permTok, float* slotW, int* slotOf) {
  int i = blockIdx.x * blockDim.x + threadIdx.x;
  if (i >= kS * kTOPK) return;
  int e = topk_id[i];
  int slot = off[e] + posArr[i];
  permTok[slot] = i >> 1; // token index
  slotW[slot] = topk_w[i];
  slotOf[i] = slot;
}

// -------------------- MoE gate+up GEMM with SwiGLU epilogue ----------------
// wgT/wuT: bf16 [E, I, H] (transposed). 4 M-tiles (64 gathered rows) per wave.
__global__ __launch_bounds__(32)
void k_moe_gateup(const __bf16* __restrict__ tbf, const __bf16* __restrict__ wgT,
                  const __bf16* __restrict__ wuT, const int* __restrict__ permTok,
                  const int* __restrict__ counts, const int* __restrict__ off,
                  __bf16* __restrict__ act) {
  int e = blockIdx.z;
  int cnt = counts[e];
  int r0 = blockIdx.y * 64;
  if (r0 >= cnt) return;
  int lane = threadIdx.x, mrow = lane & 15, hh = lane >> 4;
  int base = off[e];
  int n0 = blockIdx.x * 16;
  const __bf16* Arow[4];
#pragma unroll
  for (int t = 0; t < 4; ++t) {
    int rr = r0 + t * 16 + mrow;
    int tok = 0;
    if (rr < cnt) tok = permTok[base + rr];
    Arow[t] = tbf + (size_t)tok * kH;
  }
  const __bf16* Bg = wgT + ((size_t)e * kI + n0 + mrow) * kH;
  const __bf16* Bu = wuT + ((size_t)e * kI + n0 + mrow) * kH;
  v8f cg[4], cu[4];
#pragma unroll
  for (int t = 0; t < 4; ++t) { cg[t] = vzero8(); cu[t] = vzero8(); }
  for (int k0 = 0; k0 < kH; k0 += 32) {
    __builtin_prefetch((const void*)(Bg + k0 + 512), 0, 1);
    __builtin_prefetch((const void*)(Bu + k0 + 512), 0, 1);
    v16bf bg = load_b_contig(Bg + k0 + 16 * hh);
    v16bf bu = load_b_contig(Bu + k0 + 16 * hh);
#pragma unroll
    for (int t = 0; t < 4; ++t) {
      v16bf a = load_a_bf(Arow[t], k0, hh);
      cg[t] = wmma_bf16(a, bg, cg[t]);
      cu[t] = wmma_bf16(a, bu, cu[t]);
    }
  }
#pragma unroll
  for (int t = 0; t < 4; ++t)
#pragma unroll
    for (int v = 0; v < 8; ++v) {
      int rr = r0 + t * 16 + v + 8 * hh;
      if (rr < cnt) {
        float g = cg[t][v], u = cu[t][v];
        float sg = g / (1.f + expf(-g));
        act[(size_t)(base + rr) * kI + n0 + mrow] = (__bf16)(sg * u);
      }
    }
}

// --------------------------- MoE down GEMM ---------------------------------
// wdT: bf16 [E, H, I] (transposed). 4 M-tiles per wave; rows contiguous slots.
__global__ __launch_bounds__(32)
void k_moe_down(const __bf16* __restrict__ act, const __bf16* __restrict__ wdT,
                const int* __restrict__ counts, const int* __restrict__ off,
                const float* __restrict__ slotW, float* __restrict__ eo) {
  int e = blockIdx.z;
  int cnt = counts[e];
  int r0 = blockIdx.y * 64;
  if (r0 >= cnt) return;
  int lane = threadIdx.x, mrow = lane & 15, hh = lane >> 4;
  int base = off[e];
  int n0 = blockIdx.x * 16;
  const __bf16* Arow[4];
#pragma unroll
  for (int t = 0; t < 4; ++t) {
    int arow = base + r0 + t * 16 + mrow;
    if (arow > kS * kTOPK - 1) arow = kS * kTOPK - 1; // clamp OOB tail reads
    Arow[t] = act + (size_t)arow * kI;
  }
  const __bf16* Bp = wdT + ((size_t)e * kH + n0 + mrow) * kI;
  v8f c[4];
#pragma unroll
  for (int t = 0; t < 4; ++t) c[t] = vzero8();
  for (int k0 = 0; k0 < kI; k0 += 32) {
    __builtin_prefetch((const void*)(Bp + k0 + 512), 0, 1);
    v16bf b = load_b_contig(Bp + k0 + 16 * hh);
#pragma unroll
    for (int t = 0; t < 4; ++t) {
      v16bf a = load_a_bf(Arow[t], k0, hh);
      c[t] = wmma_bf16(a, b, c[t]);
    }
  }
#pragma unroll
  for (int t = 0; t < 4; ++t)
#pragma unroll
    for (int v = 0; v < 8; ++v) {
      int rr = r0 + t * 16 + v + 8 * hh;
      if (rr < cnt) {
        int slot = base + rr;
        eo[(size_t)slot * kH + n0 + mrow] = slotW[slot] * c[t][v];
      }
    }
}

// --------------------------- Final combine ---------------------------------
__global__ void k_combine(const float* __restrict__ res, const float* __restrict__ eo,
                          const int* __restrict__ slotOf, float* __restrict__ out) {
  int i = blockIdx.x * 256 + threadIdx.x;
  int t = i >> 11; // /H (2048)
  int h = i & (kH - 1);
  out[i] = res[i] + eo[(size_t)slotOf[2 * t] * kH + h] + eo[(size_t)slotOf[2 * t + 1] * kH + h];
}

// ---------------------------------------------------------------------------
extern "C" void kernel_launch(void* const* d_in, const int* in_sizes, int n_in,
                              void* d_out, int out_size, void* d_ws, size_t ws_size,
                              hipStream_t stream) {
  (void)in_sizes; (void)n_in; (void)out_size; (void)ws_size;
  const float* x        = (const float*)d_in[0];
  const int*   pos      = (const int*)d_in[1];
  const float* ln1_w    = (const float*)d_in[2];
  const float* q_norm_w = (const float*)d_in[3];
  const float* k_norm_w = (const float*)d_in[4];
  const float* wq       = (const float*)d_in[5];
  const float* wk       = (const float*)d_in[6];
  const float* wv       = (const float*)d_in[7];
  const float* wo       = (const float*)d_in[8];
  const float* ln2_w    = (const float*)d_in[9];
  const float* router_w = (const float*)d_in[10];
  const float* router_b = (const float*)d_in[11];
  const float* w_gate   = (const float*)d_in[12];
  const float* w_up     = (const float*)d_in[13];
  const float* w_down   = (const float*)d_in[14];
  float* out = (float*)d_out;

  char* ws = (char*)d_ws;
  size_t off_b = 0;
  auto alloc = [&](size_t bytes) -> char* {
    char* p = ws + off_b;
    off_b += (bytes + 255) & ~(size_t)255;
    return p;
  };
  // activations / intermediates
  __bf16* h_bf    = (__bf16*)alloc((size_t)kS * kH * 2);
  float*  Qf      = (float*) alloc((size_t)kS * kNH * kHD * 4);
  float*  Kf      = (float*) alloc((size_t)kS * kNKV * kHD * 4);
  float*  Vf      = (float*) alloc((size_t)kS * kNKV * kHD * 4);
  __bf16* q_bf    = (__bf16*)alloc((size_t)kNH * kS * kHD * 2);
  __bf16* k_bf    = (__bf16*)alloc((size_t)kNKV * kS * kHD * 2);
  __bf16* v_t     = (__bf16*)alloc((size_t)kNKV * kHD * kS * 2);
  __bf16* attn_bf = (__bf16*)alloc((size_t)kS * kNH * kHD * 2);
  float*  res_f   = (float*) alloc((size_t)kS * kH * 4);
  float*  t_f     = (float*) alloc((size_t)kS * kH * 4);
  __bf16* t_bf    = (__bf16*)alloc((size_t)kS * kH * 2);
  __bf16* act_bf  = (__bf16*)alloc((size_t)kS * kTOPK * kI * 2);
  float*  eo_f    = (float*) alloc((size_t)kS * kTOPK * kH * 4);
  // router / gather
  int*    topk_id = (int*)   alloc((size_t)kS * kTOPK * 4);
  float*  topk_w  = (float*) alloc((size_t)kS * kTOPK * 4);
  int*    counts  = (int*)   alloc(64);
  int*    offs    = (int*)   alloc(64);
  int*    posArr  = (int*)   alloc((size_t)kS * kTOPK * 4);
  int*    permTok = (int*)   alloc((size_t)kS * kTOPK * 4);
  float*  slotW   = (float*) alloc((size_t)kS * kTOPK * 4);
  int*    slotOf  = (int*)   alloc((size_t)kS * kTOPK * 4);
  // bf16 transposed weights [N, K]
  __bf16* wqT = (__bf16*)alloc((size_t)kH * (kNH * kHD) * 2);
  __bf16* wkT = (__bf16*)alloc((size_t)kH * (kNKV * kHD) * 2);
  __bf16* wvT = (__bf16*)alloc((size_t)kH * (kNKV * kHD) * 2);
  __bf16* woT = (__bf16*)alloc((size_t)(kNH * kHD) * kH * 2);
  __bf16* wgT = (__bf16*)alloc((size_t)kE * kH * kI * 2);
  __bf16* wuT = (__bf16*)alloc((size_t)kE * kH * kI * 2);
  __bf16* wdT = (__bf16*)alloc((size_t)kE * kI * kH * 2);

  dim3 tb(32, 8);
  // 0) one-shot weight conversion (bf16, transposed to [N,K])
  k_wt<<<dim3(kNH * kHD / 32, kH / 32, 1), tb, 0, stream>>>(wq, wqT, kH, kNH * kHD);
  k_wt<<<dim3(kNKV * kHD / 32, kH / 32, 1), tb, 0, stream>>>(wk, wkT, kH, kNKV * kHD);
  k_wt<<<dim3(kNKV * kHD / 32, kH / 32, 1), tb, 0, stream>>>(wv, wvT, kH, kNKV * kHD);
  k_wt<<<dim3(kH / 32, (kNH * kHD) / 32, 1), tb, 0, stream>>>(wo, woT, kNH * kHD, kH);
  k_wt<<<dim3(kI / 32, kH / 32, kE), tb, 0, stream>>>(w_gate, wgT, kH, kI);
  k_wt<<<dim3(kI / 32, kH / 32, kE), tb, 0, stream>>>(w_up, wuT, kH, kI);
  k_wt<<<dim3(kH / 32, kI / 32, kE), tb, 0, stream>>>(w_down, wdT, kI, kH);

  // 1) ln1 -> bf16 activations
  k_rmsnorm<<<kS, 256, 0, stream>>>(x, ln1_w, nullptr, h_bf, kH);
  // 2) QKV projections (WMMA, 64 rows per wave)
  k_gemm_t<4, false><<<dim3(kNH * kHD / 16, kS / 64), 32, 0, stream>>>(h_bf, wqT, nullptr, Qf, kNH * kHD, kH);
  k_gemm_t<4, false><<<dim3(kNKV * kHD / 16, kS / 64), 32, 0, stream>>>(h_bf, wkT, nullptr, Kf, kNKV * kHD, kH);
  k_gemm_t<4, false><<<dim3(kNKV * kHD / 16, kS / 64), 32, 0, stream>>>(h_bf, wvT, nullptr, Vf, kNKV * kHD, kH);
  // 3) q/k norms (in-place f32)
  k_rmsnorm<<<kS, 256, 0, stream>>>(Qf, q_norm_w, Qf, nullptr, kNH * kHD);
  k_rmsnorm<<<kS, 256, 0, stream>>>(Kf, k_norm_w, Kf, nullptr, kNKV * kHD);
  // 4) RoPE + head-major bf16 (V transposed for PV fragments)
  k_rope<<<dim3(kS, kNH), kHD, 0, stream>>>(Qf, pos, q_bf, kNH);
  k_rope<<<dim3(kS, kNKV), kHD, 0, stream>>>(Kf, pos, k_bf, kNKV);
  k_vt<<<dim3(kS, kNKV), kHD, 0, stream>>>(Vf, v_t);
  // 5) causal flash attention (WMMA QK^T and PV)
  k_attn<<<dim3(kS / 16, kNH), 32, 0, stream>>>(q_bf, k_bf, v_t, attn_bf);
  // 6) O-proj + residual
  k_gemm_t<4, true><<<dim3(kH / 16, kS / 64), 32, 0, stream>>>(attn_bf, woT, x, res_f, kH, kNH * kHD);
  // 7) ln2 -> router input (f32) + MoE activations (bf16)
  k_rmsnorm<<<kS, 256, 0, stream>>>(res_f, ln2_w, t_f, t_bf, kH);
  // 8) sigmoid router, top-2 with bias
  k_router<<<kS, 256, 0, stream>>>(t_f, router_w, router_b, topk_id, topk_w);
  // 9) expert gather
  k_zero_counts<<<1, 32, 0, stream>>>(counts);
  k_assign<<<(kS * kTOPK) / 256, 256, 0, stream>>>(topk_id, counts, posArr);
  k_offsets<<<1, 1, 0, stream>>>(counts, offs);
  k_scatter<<<(kS * kTOPK) / 256, 256, 0, stream>>>(topk_id, topk_w, posArr, offs, permTok, slotW, slotOf);
  // 10) expert gate/up GEMM + SwiGLU, 11) down GEMM (weighted), 12) combine
  k_moe_gateup<<<dim3(kI / 16, (kS * kTOPK) / 64, kE), 32, 0, stream>>>(t_bf, wgT, wuT, permTok, counts, offs, act_bf);
  k_moe_down<<<dim3(kH / 16, (kS * kTOPK) / 64, kE), 32, 0, stream>>>(act_bf, wdT, counts, offs, slotW, eo_f);
  k_combine<<<(kS * kH) / 256, 256, 0, stream>>>(res_f, eo_f, slotOf, out);
}
